// AttentionHead_40175124087097
// MI455X (gfx1250) — matrix operand order repaired and checked by reading, not compiled, not run on hardware
//
#include <hip/hip_runtime.h>
#include <math.h>

typedef __attribute__((ext_vector_type(16))) __bf16 v16bf;
typedef __attribute__((ext_vector_type(8)))  float  v8f;

#define E_DIM 1024
#define H_DIM 64
#define B_DIM 8
#define T_DIM 2048
#define BT    (B_DIM * T_DIM)

// ---- bf16 helpers -----------------------------------------------------------
// RNE round both, then v_perm_b32 packs the two high halves in one op.
__device__ __forceinline__ unsigned pack2bf(float a, float b) {
    unsigned ua = __builtin_bit_cast(unsigned, a);
    unsigned ub = __builtin_bit_cast(unsigned, b);
    ua += 0x7FFFu + ((ua >> 16) & 1u);
    ub += 0x7FFFu + ((ub >> 16) & 1u);
    return __builtin_amdgcn_perm(ub, ua, 0x07060302u);
}
__device__ __forceinline__ unsigned short f2bfu(float f) {
    unsigned u = __builtin_bit_cast(unsigned, f);
    u += 0x7FFFu + ((u >> 16) & 1u);
    return (unsigned short)(u >> 16);
}
// 16 contiguous bf16 -> fragment (two b128 loads)
__device__ __forceinline__ v16bf ldfrag(const unsigned short* p) {
    union { uint4 u[2]; v16bf v; } t;
    t.u[0] = *(const uint4*)p;
    t.u[1] = *(const uint4*)(p + 8);
    return t.v;
}
// two separated 8-element (16B) runs -> fragment
__device__ __forceinline__ v16bf ldfrag2(const unsigned short* p0,
                                         const unsigned short* p1) {
    union { uint4 u[2]; v16bf v; } t;
    t.u[0] = *(const uint4*)p0;
    t.u[1] = *(const uint4*)p1;
    return t.v;
}

// ---------------------------------------------------------------------------
// Kernel 0: convert Wq/Wk/Wv (f32 [E,H]) -> bf16 transposed [3][H,E]
// ---------------------------------------------------------------------------
__global__ __launch_bounds__(256)
void wconv_kernel(const float* __restrict__ Wq, const float* __restrict__ Wk,
                  const float* __restrict__ Wv, unsigned short* __restrict__ Wt) {
    int tid = blockIdx.x * 256 + threadIdx.x;          // 3*E*H = 196608 total
    int mat = tid / (E_DIM * H_DIM);
    int rem = tid % (E_DIM * H_DIM);
    int h = rem / E_DIM;
    int e = rem % E_DIM;
    const float* W = (mat == 0) ? Wq : (mat == 1) ? Wk : Wv;
    Wt[(size_t)mat * H_DIM * E_DIM + (size_t)h * E_DIM + e] =
        f2bfu(W[(size_t)e * H_DIM + h]);
}

// ---------------------------------------------------------------------------
// Kernel 1: projections, two-stage (ping-pong) software-pipelined k-loop.
// Q,K stored [BT,H]; V stored transposed [B][H][T].
// ---------------------------------------------------------------------------
__global__ __launch_bounds__(256)
void proj_kernel(const float* __restrict__ x, const unsigned short* __restrict__ Wt,
                 unsigned short* __restrict__ Qo, unsigned short* __restrict__ Ko,
                 unsigned short* __restrict__ Vt) {
    const int lane = threadIdx.x & 31;
    const int wid  = blockIdx.x * 8 + (threadIdx.x >> 5);
    const int proj = wid >> 10;                 // 0=Q 1=K 2=V
    const int tile = wid & 1023;

    const unsigned short* W = Wt + (size_t)proj * H_DIM * E_DIM;   // [H][E] bf16

    const int m  = lane & 15;
    const int kh = lane >> 4;
    const float* xrow = x + (size_t)(tile * 16 + m) * E_DIM;
    const float* xp0 = xrow + 16 * kh;         // A elements 0..7
    const float* xp1 = xrow + 16 * kh + 8;     // A elements 8..15

    const unsigned short* wp0 = W + (size_t)(0 * 16 + m) * E_DIM + 16 * kh;
    const unsigned short* wp1 = W + (size_t)(1 * 16 + m) * E_DIM + 16 * kh;
    const unsigned short* wp2 = W + (size_t)(2 * 16 + m) * E_DIM + 16 * kh;
    const unsigned short* wp3 = W + (size_t)(3 * 16 + m) * E_DIM + 16 * kh;

    v8f acc[4] = {};

#define LOAD_STAGE(F0, F1, F2, F3, W0, W1, W2, W3, KB)                         \
    F0 = *(const float4*)(xp0 + (KB));                                         \
    F1 = *(const float4*)(xp0 + (KB) + 4);                                     \
    F2 = *(const float4*)(xp1 + (KB));                                         \
    F3 = *(const float4*)(xp1 + (KB) + 4);                                     \
    W0 = ldfrag(wp0 + (KB)); W1 = ldfrag(wp1 + (KB));                          \
    W2 = ldfrag(wp2 + (KB)); W3 = ldfrag(wp3 + (KB));

#define COMPUTE_STAGE(F0, F1, F2, F3, W0, W1, W2, W3)                          \
    {                                                                          \
        union { unsigned u[8]; v16bf v; } af;                                  \
        af.u[0] = pack2bf(F0.x, F0.y); af.u[1] = pack2bf(F0.z, F0.w);          \
        af.u[2] = pack2bf(F1.x, F1.y); af.u[3] = pack2bf(F1.z, F1.w);          \
        af.u[4] = pack2bf(F2.x, F2.y); af.u[5] = pack2bf(F2.z, F2.w);          \
        af.u[6] = pack2bf(F3.x, F3.y); af.u[7] = pack2bf(F3.z, F3.w);          \
        acc[0] = __builtin_amdgcn_wmma_f32_16x16x32_bf16(                      \
            false, af.v, false, W0, (short)0, acc[0], false, false);           \
        acc[1] = __builtin_amdgcn_wmma_f32_16x16x32_bf16(                      \
            false, af.v, false, W1, (short)0, acc[1], false, false);           \
        acc[2] = __builtin_amdgcn_wmma_f32_16x16x32_bf16(                      \
            false, af.v, false, W2, (short)0, acc[2], false, false);           \
        acc[3] = __builtin_amdgcn_wmma_f32_16x16x32_bf16(                      \
            false, af.v, false, W3, (short)0, acc[3], false, false);           \
    }

    float4 fa0, fa1, fa2, fa3;  v16bf wa0, wa1, wa2, wa3;   // stage A
    float4 fb0, fb1, fb2, fb3;  v16bf wb0, wb1, wb2, wb3;   // stage B

    LOAD_STAGE(fa0, fa1, fa2, fa3, wa0, wa1, wa2, wa3, 0)

    for (int kb = 0; kb < E_DIM; kb += 64) {
        LOAD_STAGE(fb0, fb1, fb2, fb3, wb0, wb1, wb2, wb3, kb + 32)   // always valid
        COMPUTE_STAGE(fa0, fa1, fa2, fa3, wa0, wa1, wa2, wa3)
        if (kb + 64 < E_DIM) {
            LOAD_STAGE(fa0, fa1, fa2, fa3, wa0, wa1, wa2, wa3, kb + 64)
        }
        COMPUTE_STAGE(fb0, fb1, fb2, fb3, wb0, wb1, wb2, wb3)
    }
#undef LOAD_STAGE
#undef COMPUTE_STAGE

    // C layout: element e -> row e+8*kh, col s*16+m
    if (proj < 2) {
        unsigned short* Out = (proj == 0) ? Qo : Ko;
        #pragma unroll
        for (int s = 0; s < 4; ++s)
            #pragma unroll
            for (int e = 0; e < 8; ++e)
                Out[(size_t)(tile * 16 + e + 8 * kh) * H_DIM + s * 16 + m] =
                    f2bfu(acc[s][e]);
    } else {
        #pragma unroll
        for (int s = 0; s < 4; ++s) {
            const int h = s * 16 + m;
            #pragma unroll
            for (int e = 0; e < 8; ++e) {
                const int row = tile * 16 + e + 8 * kh;   // b*T + t
                const int b = row >> 11, t = row & (T_DIM - 1);
                Vt[((size_t)b * H_DIM + h) * T_DIM + t] = f2bfu(acc[s][e]);
            }
        }
    }
}

// ---------------------------------------------------------------------------
// Kernel 2: causal flash attention. Single-wave workgroups (32 thr); each wave
// owns 16 Q rows, iterates 64-wide key tiles. 1024 blocks -> full WGP spread.
// ---------------------------------------------------------------------------
__global__ __launch_bounds__(32)
void attn_kernel(const unsigned short* __restrict__ Qb,
                 const unsigned short* __restrict__ Kb,
                 const unsigned short* __restrict__ Vtb,
                 float* __restrict__ out) {
    __shared__ unsigned short ldsP[16][64];    // [row][key] bf16 bits

    const int lane  = threadIdx.x;             // 0..31
    const int b     = blockIdx.x >> 7;         // 128 q-tiles per batch
    const int qt    = blockIdx.x & 127;
    const int qbase = qt * 16;
    const int m     = lane & 15;
    const int kh    = lane >> 4;

    const unsigned short* Q  = Qb  + (size_t)b * T_DIM * H_DIM;
    const unsigned short* K  = Kb  + (size_t)b * T_DIM * H_DIM;
    const unsigned short* Vt = Vtb + (size_t)b * H_DIM * T_DIM;   // [H][T]

    // Q tile 16x64: two A-frags
    v16bf qf[2];
    {
        const unsigned short* qrow = Q + (size_t)(qbase + m) * H_DIM;
        qf[0] = ldfrag(qrow + 16 * kh);
        qf[1] = ldfrag(qrow + 32 + 16 * kh);
    }

    float mrow[8], lrow[8];
    v8f o[4] = {};
    #pragma unroll
    for (int e = 0; e < 8; ++e) { mrow[e] = -__builtin_inff(); lrow[e] = 0.f; }

    const int ktiles = (qbase + 15) / 64 + 1;

    for (int kt = 0; kt < ktiles; ++kt) {
        const int kbase = kt * 64;

        if (kt + 1 < ktiles) {               // prefetch next tile (K rows + Vt runs)
            const unsigned short* pk = K + (size_t)(kbase + 64 + lane) * H_DIM;
            __builtin_prefetch(pk, 0, 1);
            __builtin_prefetch(pk + 32 * H_DIM, 0, 1);
            __builtin_prefetch(Vt + (size_t)lane * T_DIM + kbase + 64, 0, 1);
            __builtin_prefetch(Vt + (size_t)(lane + 32) * T_DIM + kbase + 64, 0, 1);
        }

        // ---- S = Q K^T : load all 8 K-frags, then 8 WMMAs ----
        v16bf kfr[4][2];
        #pragma unroll
        for (int s = 0; s < 4; ++s) {
            const unsigned short* krow =
                K + (size_t)(kbase + s * 16 + m) * H_DIM + 16 * kh;
            kfr[s][0] = ldfrag(krow);        // h = 16*kh + e
            kfr[s][1] = ldfrag(krow + 32);   // h = 32 + 16*kh + e
        }
        v8f sacc[4];
        #pragma unroll
        for (int s = 0; s < 4; ++s) {
            v8f z = {};
            z = __builtin_amdgcn_wmma_f32_16x16x32_bf16(
                false, qf[0], false, kfr[s][0], (short)0, z, false, false);
            sacc[s] = __builtin_amdgcn_wmma_f32_16x16x32_bf16(
                false, qf[1], false, kfr[s][1], (short)0, z, false, false);
        }

        // ---- scale, causal mask, online softmax ----
        float pv[4][8], curmax[8];
        #pragma unroll
        for (int e = 0; e < 8; ++e) {
            const int row = qbase + e + 8 * kh;
            float mx = -__builtin_inff();
            #pragma unroll
            for (int s = 0; s < 4; ++s) {
                float v = sacc[s][e] * 0.125f;          // 1/sqrt(64)
                if (kbase + s * 16 + m > row) v = -__builtin_inff();
                pv[s][e] = v;
                mx = fmaxf(mx, v);
            }
            #pragma unroll
            for (int off = 1; off < 16; off <<= 1)
                mx = fmaxf(mx, __shfl_xor(mx, off, 32));
            curmax[e] = mx;
        }
        #pragma unroll
        for (int e = 0; e < 8; ++e) {
            float mnew  = fmaxf(mrow[e], curmax[e]);
            float alpha = __expf(mrow[e] - mnew);       // 0 on first tile
            float ls = 0.f;
            #pragma unroll
            for (int s = 0; s < 4; ++s) {
                float p = __expf(pv[s][e] - mnew);
                pv[s][e] = p;
                ls += p;
            }
            #pragma unroll
            for (int off = 1; off < 16; off <<= 1)
                ls += __shfl_xor(ls, off, 32);
            lrow[e] = lrow[e] * alpha + ls;
            mrow[e] = mnew;
            #pragma unroll
            for (int s = 0; s < 4; ++s) o[s][e] *= alpha;
        }

        // ---- issue V-frag loads early: overlap with P LDS round-trip ----
        v16bf vfr[4][2];
        #pragma unroll
        for (int s = 0; s < 4; ++s) {
            const unsigned short* vrow = Vt + (size_t)(s * 16 + m) * T_DIM + kbase;
            vfr[s][0] = ldfrag(vrow + 16 * kh);         // keys kbase+0..31
            vfr[s][1] = ldfrag(vrow + 32 + 16 * kh);    // keys kbase+32..63
        }

        // ---- P (16x64): C-layout -> A-layout via LDS ----
        #pragma unroll
        for (int e = 0; e < 8; ++e)
            #pragma unroll
            for (int s = 0; s < 4; ++s)
                ldsP[e + 8 * kh][s * 16 + m] = f2bfu(pv[s][e]);

        __builtin_amdgcn_wave_barrier();
        asm volatile("s_wait_dscnt 0" ::: "memory");

        v16bf pf0 = ldfrag(&ldsP[m][16 * kh]);
        v16bf pf1 = ldfrag(&ldsP[m][32 + 16 * kh]);
        __builtin_amdgcn_wave_barrier();

        // ---- O += P V ----
        #pragma unroll
        for (int s = 0; s < 4; ++s) {
            o[s] = __builtin_amdgcn_wmma_f32_16x16x32_bf16(
                false, pf0, false, vfr[s][0], (short)0, o[s], false, false);
            o[s] = __builtin_amdgcn_wmma_f32_16x16x32_bf16(
                false, pf1, false, vfr[s][1], (short)0, o[s], false, false);
        }
    }

    // ---- epilogue ----
    float* orow = out + (size_t)(b * T_DIM + qbase) * H_DIM;
    #pragma unroll
    for (int e = 0; e < 8; ++e) {
        const float inv = 1.0f / lrow[e];
        #pragma unroll
        for (int s = 0; s < 4; ++s)
            orow[(size_t)(e + 8 * kh) * H_DIM + s * 16 + m] = o[s][e] * inv;
    }
}

// ---------------------------------------------------------------------------
extern "C" void kernel_launch(void* const* d_in, const int* in_sizes, int n_in,
                              void* d_out, int out_size, void* d_ws, size_t ws_size,
                              hipStream_t stream) {
    const float* x  = (const float*)d_in[0];
    const float* Wq = (const float*)d_in[1];
    const float* Wk = (const float*)d_in[2];
    const float* Wv = (const float*)d_in[3];

    unsigned short* Qw  = (unsigned short*)d_ws;               // [BT,H]
    unsigned short* Kw  = Qw + (size_t)BT * H_DIM;             // [BT,H]
    unsigned short* Vtw = Kw + (size_t)BT * H_DIM;             // [B][H][T]
    unsigned short* Wt  = Vtw + (size_t)BT * H_DIM;            // [3][H,E]

    wconv_kernel<<<768, 256, 0, stream>>>(Wq, Wk, Wv, Wt);
    proj_kernel<<<384, 256, 0, stream>>>(x, Wt, Qw, Kw, Vtw);
    attn_kernel<<<1024, 32, 0, stream>>>(Qw, Kw, Vtw, (float*)d_out);
}
